// GatedMultiScaleRetention_29248727286037
// MI455X (gfx1250) — matrix operand order repaired and checked by reading, not compiled, not run on hardware
//
#include <hip/hip_runtime.h>

#define SEQ 2048
#define DM 2048
#define NH 16
#define DH 128
#define K3 6144

typedef __bf16 bf16;
typedef __attribute__((ext_vector_type(8)))  __bf16 v8bf;
typedef __attribute__((ext_vector_type(16))) __bf16 v16bf;
typedef __attribute__((ext_vector_type(8)))  float  v8f;
typedef __attribute__((ext_vector_type(4)))  unsigned int u32x4;
typedef __attribute__((ext_vector_type(8)))  int i32x8;
typedef __attribute__((ext_vector_type(4)))  int i32x4;

#define DEV __device__ __forceinline__

// ---- WMMA fragment loaders (contiguous k order k = 16*(lane>>4) + s, wave32) ----

// A-matrix 16x32 bf16: lane m=lane&15 holds row m; lanes 0-15 K=[k0..k0+15];
// lanes 16-31 K=[k0+16..k0+31].
DEV v16bf load_a_frag(const bf16* base, int ld, int row0, int k0, int lane) {
  int m  = lane & 15;
  int ko = (lane >> 4) << 4;
  const bf16* p = base + (size_t)(row0 + m) * ld + k0 + ko;
  v8bf lo = *(const v8bf*)p;
  v8bf hi = *(const v8bf*)(p + 8);
  v16bf r;
#pragma unroll
  for (int i = 0; i < 8; ++i) { r[i] = lo[i]; r[i + 8] = hi[i]; }
  return r;
}

// B-matrix 32x16 bf16 from a row-major (N x K) "Bt" buffer (C = A @ Bt^T):
// lane n=lane&15 holds column n; lanes 0-15 K=[k0..k0+15], lanes 16-31 K=[k0+16..k0+31].
DEV v16bf load_b_frag(const bf16* base, int ld, int col0, int k0, int lane) {
  int n  = lane & 15;
  int kb = (lane >> 4) << 4;
  return *(const v16bf*)(base + (size_t)(col0 + n) * ld + k0 + kb);
}

DEV v8f wmma_bf16(v16bf a, v16bf b, v8f c) {
  return __builtin_amdgcn_wmma_f32_16x16x32_bf16(false, a, false, b, (short)0, c,
                                                 false, false);
}

// ---- Tensor Data Mover: stage a 2-D bf16 tile (tile_rows x tile_cols) into LDS ----
// D# per CDNA5 ISA 8.3/8.4: group0 = {count=1, lds_addr, global_addr, type=2},
// group1 = {data_size=2B, tensor_dim0/1, tile_dim0/1, dim0_stride}; groups 2/3 zero (2-D).
// This toolchain's builtin takes 6 args (g0, g1, g2, g3, extra, cpol).
DEV void tdm_load_tile_bf16(unsigned lds_off, const bf16* gsrc,
                            unsigned tile_cols, unsigned tile_rows,
                            unsigned tensor_cols, unsigned tensor_rows,
                            unsigned ld_elems) {
  unsigned long long ga = (unsigned long long)(size_t)gsrc;
  u32x4 g0;
  g0[0] = 1u;                                       // count=1 (valid user descriptor)
  g0[1] = lds_off;                                  // LDS byte address
  g0[2] = (unsigned)(ga & 0xffffffffu);             // global_addr[31:0]
  g0[3] = (unsigned)((ga >> 32) & 0x01ffffffu) | (2u << 30);  // addr[56:32] | type=2
  unsigned long long s0 = ld_elems;                 // tensor_dim0_stride (elements)
  i32x8 g1;
  g1[0] = (int)(1u << 16);                          // data_size=1 -> 2 bytes; no mask/pad
  g1[1] = (int)((tensor_cols & 0xffffu) << 16);     // tensor_dim0[15:0] at bits 63:48
  g1[2] = (int)(((tensor_cols >> 16) & 0xffffu) | ((tensor_rows & 0xffffu) << 16));
  g1[3] = (int)(((tensor_rows >> 16) & 0xffffu) | ((tile_cols & 0xffffu) << 16));
  g1[4] = (int)(tile_rows & 0xffffu);               // tile_dim1; tile_dim2=0 (unused)
  g1[5] = (int)(unsigned)(s0 & 0xffffffffu);        // dim0_stride[31:0]
  g1[6] = (int)(unsigned)((s0 >> 32) & 0xffffu);    // dim0_stride[47:32]; dim1_stride=0
  g1[7] = 0;
  i32x4 z4 = {};
  i32x8 z8 = {};
  __builtin_amdgcn_tensor_load_to_lds(g0, g1, z4, z4, z8, 0);
}

// D/C f32 16x16 layout: element r of accum -> row = r + 8*(lane>>4), col = lane&15.

// ---------------- elementwise fp32 -> bf16 ----------------
__global__ void f2bf_kernel(const float* __restrict__ src, bf16* __restrict__ dst, int n) {
  int i = blockIdx.x * blockDim.x + threadIdx.x;
  if (i < n) dst[i] = (bf16)src[i];
}

// ---- shared GEMM core: A panel (16 x DM) TDM-staged in LDS, each wave does 16x32 ----
DEV void gemm_core(const bf16* __restrict__ A, const bf16* __restrict__ B,
                   bf16* Alds, int row0, int col0, int lane, v8f& acc0, v8f& acc1) {
  if (threadIdx.x < 32) {
    tdm_load_tile_bf16((unsigned)(size_t)Alds, A + (size_t)row0 * DM,
                       DM, 16, DM, (unsigned)(SEQ - row0), DM);
    __builtin_amdgcn_s_wait_tensorcnt(0);
  }
  __syncthreads();
  for (int k0 = 0; k0 < DM; k0 += 32) {
    v16bf a  = load_a_frag(Alds, DM, 0, k0, lane);       // ds_load from TDM-staged panel
    v16bf b0 = load_b_frag(B, DM, col0, k0, lane);
    v16bf b1 = load_b_frag(B, DM, col0 + 16, k0, lane);
    acc0 = wmma_bf16(a, b0, acc0);                       // A frag reused 2x
    acc1 = wmma_bf16(a, b1, acc1);
  }
}

// ---------------- qkv GEMM: C(2048x6144) = Xbf @ Wqkvbf^T ----------------
__global__ void gemm_qkv_kernel(const bf16* __restrict__ A, const bf16* __restrict__ B,
                                float* __restrict__ C) {
  __shared__ alignas(16) bf16 Alds[16 * DM];
  int lane = threadIdx.x & 31, w = threadIdx.x >> 5;
  int row0 = blockIdx.x * 16;
  int col0 = (blockIdx.y * 8 + w) * 32;
  v8f acc0 = {}, acc1 = {};
  gemm_core(A, B, Alds, row0, col0, lane, acc0, acc1);
#pragma unroll
  for (int r = 0; r < 8; ++r) {
    size_t row = row0 + r + ((lane >> 4) << 3);
    C[row * K3 + col0 + (lane & 15)]      = acc0[r];
    C[row * K3 + col0 + 16 + (lane & 15)] = acc1[r];
  }
}

// ---------------- alphas: -log1p(exp(-(x @ Walpha^T))) ----------------
__global__ void alphas_kernel(const float* __restrict__ x, const float* __restrict__ wa,
                              float* __restrict__ alphas) {
  __shared__ float red[256];
  int i = blockIdx.x;
  float s = 0.f;
  for (int c = threadIdx.x; c < DM; c += 256) s += x[(size_t)i * DM + c] * wa[c];
  red[threadIdx.x] = s;
  __syncthreads();
  for (int off = 128; off; off >>= 1) {
    if ((int)threadIdx.x < off) red[threadIdx.x] += red[threadIdx.x + off];
    __syncthreads();
  }
  if (threadIdx.x == 0) {
    float z = red[0];
    alphas[i] = -log1pf(__expf(-z));
  }
}

// ---------------- serial prefix sum over 2048 scalars ----------------
__global__ void cumsum_kernel(const float* __restrict__ alphas, float* __restrict__ a) {
  if (threadIdx.x == 0 && blockIdx.x == 0) {
    float s = 0.f;
    for (int i = 0; i < SEQ; ++i) { s += alphas[i]; a[i] = s; }
  }
}

// ---------------- xpos rotation + k decay + v transpose, fp32 -> bf16 ----------------
__global__ void xpos_kernel(const float* __restrict__ qkv, const float* __restrict__ alphas,
                            bf16* __restrict__ qbf, bf16* __restrict__ kbf,
                            bf16* __restrict__ vtbf) {
  int i = blockIdx.x, h = blockIdx.y, d = threadIdx.x;  // d in [0,64)
  const float* row = qkv + (size_t)i * K3;
  float f   = __expf(-(float)d * (9.210340371976184f / 64.f));  // 10000^(-d/64)
  float ang = (float)i * f;
  float sn = __sinf(ang), cs = __cosf(ang);
  float sbase = ((float)d + 0.4f * 128.f) / (1.4f * 128.f);
  float sc  = __powf(sbase, (float)i / 512.f);
  float isc = 1.f / sc;

  float x1 = row[h * DH + d], x2 = row[h * DH + d + 64];
  bf16* qrow = qbf + ((size_t)h * SEQ + i) * DH;
  qrow[d]      = (bf16)((x1 * cs - x2 * sn) * sc);
  qrow[d + 64] = (bf16)((x1 * sn + x2 * cs) * sc);

  float decay = 1.f - __expf(alphas[i]);
  x1 = row[DM + h * DH + d]; x2 = row[DM + h * DH + d + 64];
  bf16* krow = kbf + ((size_t)h * SEQ + i) * DH;
  krow[d]      = (bf16)((x1 * cs - x2 * sn) * isc * decay);
  krow[d + 64] = (bf16)((x1 * sn + x2 * cs) * isc * decay);

  float v1 = row[2 * DM + h * DH + d], v2 = row[2 * DM + h * DH + d + 64];
  bf16* vh = vtbf + (size_t)h * DH * SEQ;
  vh[(size_t)d * SEQ + i]        = (bf16)v1;
  vh[(size_t)(d + 64) * SEQ + i] = (bf16)v2;
}

// ---------------- S0^T = (exp(a_last) * state)^T as bf16 ----------------
__global__ void prep_state_kernel(const float* __restrict__ state, const float* __restrict__ a,
                                  bf16* __restrict__ s0t) {
  int h = blockIdx.x, e = blockIdx.y, d = threadIdx.x;
  float g = __expf(a[SEQ - 1]);
  s0t[((size_t)h * DH + e) * DH + d] = (bf16)(g * state[((size_t)h * DH + d) * DH + e]);
}

// ---------------- flash-retention: ret[i,h,:] = sum_{j<=i} (q_i.k_j) e^{a_i-a_j} v_j + q_i @ S0
__global__ void retention_kernel(const bf16* __restrict__ qbf, const bf16* __restrict__ kbf,
                                 const bf16* __restrict__ vtbf, const bf16* __restrict__ s0t,
                                 const float* __restrict__ a, float* __restrict__ ret) {
  __shared__ alignas(16) bf16 Plds[4][16][32];
  int lane = threadIdx.x & 31, w = threadIdx.x >> 5;
  int wid = blockIdx.x * 4 + w;
  int h = wid >> 7, it = wid & 127;
  int row0 = it * 16;
  const bf16* qh  = qbf  + (size_t)h * SEQ * DH;
  const bf16* kh  = kbf  + (size_t)h * SEQ * DH;
  const bf16* vth = vtbf + (size_t)h * DH * SEQ;
  const bf16* sh  = s0t  + (size_t)h * DH * DH;
  bf16(*P)[32] = Plds[w];

  v16bf qa[4];
#pragma unroll
  for (int kk = 0; kk < 4; ++kk) qa[kk] = load_a_frag(qh, DH, row0, kk * 32, lane);

  v8f acc[8];
#pragma unroll
  for (int n = 0; n < 8; ++n) {
    v8f z = {};
    acc[n] = z;
#pragma unroll
    for (int kk = 0; kk < 4; ++kk)
      acc[n] = wmma_bf16(qa[kk], load_b_frag(sh, DH, n * 16, kk * 32, lane), acc[n]);
  }

  float ai[8];
#pragma unroll
  for (int r = 0; r < 8; ++r) ai[r] = a[row0 + r + ((lane >> 4) << 3)];

  for (int jc = 0; jc < row0 + 16; jc += 32) {
#pragma unroll
    for (int s = 0; s < 2; ++s) {
      int j = jc + s * 16;
      v8f sacc = {};
#pragma unroll
      for (int kk = 0; kk < 4; ++kk)
        sacc = wmma_bf16(qa[kk], load_b_frag(kh, DH, j, kk * 32, lane), sacc);
      float aj = a[j + (lane & 15)];
#pragma unroll
      for (int r = 0; r < 8; ++r) {
        int gi = row0 + r + ((lane >> 4) << 3);
        int gj = j + (lane & 15);
        float pv = (gj <= gi) ? sacc[r] * __expf(ai[r] - aj) : 0.f;
        P[r + ((lane >> 4) << 3)][s * 16 + (lane & 15)] = (bf16)pv;  // D-layout -> LDS
      }
    }
    // reshape through LDS into an A-fragment (in-order DS within wave; compiler waits dscnt)
    v16bf pa = load_a_frag(&P[0][0], 32, 0, 0, lane);
#pragma unroll
    for (int n = 0; n < 8; ++n)
      acc[n] = wmma_bf16(pa, load_b_frag(vth, SEQ, n * 16, jc, lane), acc[n]);
  }

#pragma unroll
  for (int n = 0; n < 8; ++n)
#pragma unroll
    for (int r = 0; r < 8; ++r)
      ret[(size_t)(row0 + r + ((lane >> 4) << 3)) * DM + h * DH + n * 16 + (lane & 15)] =
          acc[n][r];
}

// ---------------- new_state = exp(a_last)*state + sum_i exp(a[S-1-i]) k_i v_i^T ----------------
__global__ void new_state_kernel(const float* __restrict__ state, const float* __restrict__ a,
                                 const bf16* __restrict__ kbf, const bf16* __restrict__ vtbf,
                                 float* __restrict__ ns) {
  int h = blockIdx.x, d = blockIdx.y, e = threadIdx.x;
  const bf16* khd = kbf + (size_t)h * SEQ * DH + d;
  const bf16* vhe = vtbf + (size_t)h * DH * SEQ + (size_t)e * SEQ;
  float acc = 0.f;
  for (int i = 0; i < SEQ; ++i) {
    float wgt = __expf(a[SEQ - 1 - i]);
    acc += wgt * (float)khd[(size_t)i * DH] * (float)vhe[i];
  }
  float g = __expf(a[SEQ - 1]);
  size_t idx = ((size_t)h * DH + d) * DH + e;
  ns[idx] = g * state[idx] + acc;
}

// ---------------- per-(i,h) groupnorm over 128, emit fp32 + bf16 ----------------
__global__ void gn_kernel(const float* __restrict__ ret, const float* __restrict__ gw,
                          const float* __restrict__ gb, float* __restrict__ retn,
                          bf16* __restrict__ retnbf) {
  __shared__ float s1[128], s2[128];
  int i = blockIdx.x, h = blockIdx.y, d = threadIdx.x;
  size_t idx = (size_t)i * DM + h * DH + d;
  float x = ret[idx];
  s1[d] = x; s2[d] = x * x;
  __syncthreads();
  for (int off = 64; off; off >>= 1) {
    if (d < off) { s1[d] += s1[d + off]; s2[d] += s2[d + off]; }
    __syncthreads();
  }
  float mu  = s1[0] * (1.f / 128.f);
  float var = s2[0] * (1.f / 128.f) - mu * mu;
  float r = (x - mu) * rsqrtf(var + 1e-5f);
  float o = r * gw[h * DH + d] + gb[h * DH + d];
  retn[idx]  = o;
  retnbf[idx] = (bf16)o;
}

// ---------------- gate GEMM: y = swish(retn @ Wgate^T) * retn, bf16 out ----------------
__global__ void gemm_gate_kernel(const bf16* __restrict__ A, const bf16* __restrict__ B,
                                 const float* __restrict__ retn, bf16* __restrict__ ybf) {
  __shared__ alignas(16) bf16 Alds[16 * DM];
  int lane = threadIdx.x & 31, w = threadIdx.x >> 5;
  int row0 = blockIdx.x * 16;
  int col0 = (blockIdx.y * 8 + w) * 32;
  v8f acc0 = {}, acc1 = {};
  gemm_core(A, B, Alds, row0, col0, lane, acc0, acc1);
#pragma unroll
  for (int t = 0; t < 2; ++t) {
#pragma unroll
    for (int r = 0; r < 8; ++r) {
      size_t row = row0 + r + ((lane >> 4) << 3);
      int col = col0 + t * 16 + (lane & 15);
      float c = t ? acc1[r] : acc0[r];
      float g = c / (1.f + __expf(-c));  // swish
      ybf[row * DM + col] = (bf16)(g * retn[row * DM + col]);
    }
  }
}

// ---------------- out GEMM: out = y @ Wout^T, fp32 to d_out ----------------
__global__ void gemm_out_kernel(const bf16* __restrict__ A, const bf16* __restrict__ B,
                                float* __restrict__ C) {
  __shared__ alignas(16) bf16 Alds[16 * DM];
  int lane = threadIdx.x & 31, w = threadIdx.x >> 5;
  int row0 = blockIdx.x * 16;
  int col0 = (blockIdx.y * 8 + w) * 32;
  v8f acc0 = {}, acc1 = {};
  gemm_core(A, B, Alds, row0, col0, lane, acc0, acc1);
#pragma unroll
  for (int r = 0; r < 8; ++r) {
    size_t row = row0 + r + ((lane >> 4) << 3);
    C[row * DM + col0 + (lane & 15)]      = acc0[r];
    C[row * DM + col0 + 16 + (lane & 15)] = acc1[r];
  }
}

extern "C" void kernel_launch(void* const* d_in, const int* in_sizes, int n_in,
                              void* d_out, int out_size, void* d_ws, size_t ws_size,
                              hipStream_t stream) {
  (void)in_sizes; (void)n_in; (void)out_size; (void)ws_size;
  const float* x      = (const float*)d_in[0];
  const float* state  = (const float*)d_in[1];
  const float* Wqkv   = (const float*)d_in[2];
  const float* Walpha = (const float*)d_in[3];
  const float* gnw    = (const float*)d_in[4];
  const float* gnb    = (const float*)d_in[5];
  const float* Wgate  = (const float*)d_in[6];
  const float* Wout   = (const float*)d_in[7];

  char* ws = (char*)d_ws;
  size_t off = 0;
  auto alloc = [&](size_t bytes) -> char* {
    char* p = ws + off;
    off += (bytes + 255) & ~(size_t)255;
    return p;
  };
  bf16*  xbf     = (bf16*)alloc((size_t)SEQ * DM * 2);
  bf16*  wqkvbf  = (bf16*)alloc((size_t)K3 * DM * 2);
  bf16*  wgatebf = (bf16*)alloc((size_t)DM * DM * 2);
  bf16*  woutbf  = (bf16*)alloc((size_t)DM * DM * 2);
  float* qkvf    = (float*)alloc((size_t)SEQ * K3 * 4);
  bf16*  qbf     = (bf16*)alloc((size_t)SEQ * DM * 2);
  bf16*  kbf     = (bf16*)alloc((size_t)SEQ * DM * 2);
  bf16*  vtbf    = (bf16*)alloc((size_t)SEQ * DM * 2);
  float* alphas  = (float*)alloc((size_t)SEQ * 4);
  float* acum    = (float*)alloc((size_t)SEQ * 4);
  bf16*  s0t     = (bf16*)alloc((size_t)NH * DH * DH * 2);
  float* retb    = (float*)alloc((size_t)SEQ * DM * 4);
  float* retn    = (float*)alloc((size_t)SEQ * DM * 4);
  bf16*  retnbf  = (bf16*)alloc((size_t)SEQ * DM * 2);
  bf16*  ybf     = (bf16*)alloc((size_t)SEQ * DM * 2);

  float* outp = (float*)d_out;
  float* nsp  = outp + (size_t)SEQ * DM;

  // 1) bf16 staging of GEMM operands
  f2bf_kernel<<<(SEQ * DM + 255) / 256, 256, 0, stream>>>(x, xbf, SEQ * DM);
  f2bf_kernel<<<(K3 * DM + 255) / 256, 256, 0, stream>>>(Wqkv, wqkvbf, K3 * DM);
  f2bf_kernel<<<(DM * DM + 255) / 256, 256, 0, stream>>>(Wgate, wgatebf, DM * DM);
  f2bf_kernel<<<(DM * DM + 255) / 256, 256, 0, stream>>>(Wout, woutbf, DM * DM);

  // 2) qkv projection (WMMA + TDM-staged A panel)
  gemm_qkv_kernel<<<dim3(SEQ / 16, K3 / 256), 256, 0, stream>>>(xbf, wqkvbf, qkvf);

  // 3) decay logits + prefix sum
  alphas_kernel<<<SEQ, 256, 0, stream>>>(x, Walpha, alphas);
  cumsum_kernel<<<1, 32, 0, stream>>>(alphas, acum);

  // 4) xpos rotate q/k, decay k, transpose v; 5) decayed initial state
  xpos_kernel<<<dim3(SEQ, NH), 64, 0, stream>>>(qkvf, alphas, qbf, kbf, vtbf);
  prep_state_kernel<<<dim3(NH, DH), DH, 0, stream>>>(state, acum, s0t);

  // 6) retention core (WMMA flash loop)
  retention_kernel<<<(NH * (SEQ / 16)) / 4, 128, 0, stream>>>(qbf, kbf, vtbf, s0t, acum, retb);

  // 7) state update output
  new_state_kernel<<<dim3(NH, DH), DH, 0, stream>>>(state, acum, kbf, vtbf, nsp);

  // 8) groupnorm
  gn_kernel<<<dim3(SEQ, NH), DH, 0, stream>>>(retb, gnw, gnb, retn, retnbf);

  // 9) gate GEMM + swish*retn (WMMA); 10) output GEMM (WMMA)
  gemm_gate_kernel<<<dim3(SEQ / 16, DM / 256), 256, 0, stream>>>(retnbf, wgatebf, retn, ybf);
  gemm_out_kernel<<<dim3(SEQ / 16, DM / 256), 256, 0, stream>>>(ybf, woutbf, outp);
}
